// VectorQuantizerEMA_54391465837245
// MI455X (gfx1250) — compile-verified, matrix operand order checked by reading, not run on hardware
//
#include <hip/hip_runtime.h>
#include <hip/hip_bf16.h>

// ---------------------------------------------------------------------------
// VQ-VAE EMA vector quantizer for MI455X (gfx1250, wave32, WMMA + async-LDS).
//   N=16384 points, D=256 dims, K=8192 codes.
//   argmin_k ||x - e_k||^2 == argmin_k ( e_k^2 - 2 x.e_k )
//   dots via v_wmma_f32_16x16x32_bf16; B tiles staged with
//   global_load_async_to_lds_b128 double-buffering (ASYNCcnt pipeline).
// ---------------------------------------------------------------------------

typedef __attribute__((ext_vector_type(16))) __bf16        v16bf;
typedef __attribute__((ext_vector_type(8)))  float         v8f;
typedef __attribute__((ext_vector_type(8)))  unsigned int  v8u;

#define NUM_E  8192   // K
#define DIM    256    // D
#define NPTS   16384  // N
#define KSPLIT 4      // K-dimension split across blockIdx.y
#define KCODES (NUM_E / KSPLIT)   // 2048 codes per split
#define TILE_C 64                 // codes per LDS tile (32 KB per buffer)
#define TILES  (KCODES / TILE_C)  // 32 tile iterations per block

// ---------------------------------------------------------------------------
// Kernel 1: embedding f32 -> bf16 (ws), e2[k], new_ema_w init = ema_w * DECAY.
// ---------------------------------------------------------------------------
__global__ __launch_bounds__(256) void vq_prep_kernel(
    const float* __restrict__ emb, const float* __restrict__ emaw,
    __bf16* __restrict__ embB, float* __restrict__ e2,
    float* __restrict__ newemaw)
{
    __shared__ float sred[8];
    const int k = blockIdx.x;
    const int t = threadIdx.x;
    const size_t i = (size_t)k * DIM + t;
    const float v = emb[i];
    embB[i]    = (__bf16)v;
    newemaw[i] = emaw[i] * 0.99f;
    float s = v * v;
    #pragma unroll
    for (int off = 16; off > 0; off >>= 1) s += __shfl_xor(s, off, 32);
    if ((t & 31) == 0) sred[t >> 5] = s;
    __syncthreads();
    if (t == 0) {
        float tot = 0.f;
        #pragma unroll
        for (int w = 0; w < 8; ++w) tot += sred[w];
        e2[k] = tot;
    }
}

// ---------------------------------------------------------------------------
// Async stage of one 64-code (32 KB) bf16 tile: 8 b128 async copies per thread
// (ASYNCcnt += 8 per wave). Instruction offset is added to BOTH the LDS and
// the global address (ISA 08_async_tensor §4.4), and both strides are 4096 B.
// ---------------------------------------------------------------------------
__device__ __forceinline__ void stage_async_tile(const uint4* __restrict__ gsrc,
                                                 unsigned lds_byte)
{
    #pragma unroll
    for (int i = 0; i < 8; ++i) {
        asm volatile("global_load_async_to_lds_b128 %0, %1, off offset:%2"
                     :: "v"(lds_byte), "v"(gsrc), "n"(i * 4096)
                     : "memory");
    }
}

// ---------------------------------------------------------------------------
// Kernel 2: fused distance GEMM + argmin, K-split partials.
// grid = (N/128, KSPLIT), block = 256 threads = 8 waves.
// Each wave holds its 16x256 bf16 A panel in VGPRs (8 WMMA fragments) for the
// whole scan. B tiles (64 codes x 256 dims bf16) are double-buffered in LDS
// via async copies; compute overlaps the next tile's DMA.
// ---------------------------------------------------------------------------
__global__ __launch_bounds__(256) void vq_argmin_kernel(
    const float* __restrict__ x, const __bf16* __restrict__ embB,
    const float* __restrict__ e2, float* __restrict__ pvals,
    int* __restrict__ pidx)
{
    __shared__ uint4 sB[4096];                 // 2 buffers x 64 rows x 512 B
    const int t    = threadIdx.x;
    const int wave = t >> 5;
    const int lane = t & 31;
    const int half = lane >> 4;
    const int mn   = lane & 15;
    const int rowBase   = blockIdx.x * 128 + wave * 16;
    const int split     = blockIdx.y;
    const int splitBase = split * KCODES;

    // ---- load A fragments (16-bit A 16x32 layout, ISA 7.12.2) ----
    v16bf a[8];
    const float* rowp = x + (size_t)(rowBase + mn) * DIM;
    #pragma unroll
    for (int f = 0; f < 8; ++f) {
        #pragma unroll
        for (int run = 0; run < 2; ++run) {
            const float4* p = (const float4*)(rowp + f * 32 + run * 16 + half * 8);
            float4 x0 = p[0], x1 = p[1];
            a[f][run * 8 + 0] = (__bf16)x0.x;
            a[f][run * 8 + 1] = (__bf16)x0.y;
            a[f][run * 8 + 2] = (__bf16)x0.z;
            a[f][run * 8 + 3] = (__bf16)x0.w;
            a[f][run * 8 + 4] = (__bf16)x1.x;
            a[f][run * 8 + 5] = (__bf16)x1.y;
            a[f][run * 8 + 6] = (__bf16)x1.z;
            a[f][run * 8 + 7] = (__bf16)x1.w;
        }
    }

    float minV[8];
    int   minIdx[8];
    #pragma unroll
    for (int r = 0; r < 8; ++r) { minV[r] = 3.4e38f; minIdx[r] = 0; }

    const uint4* embBq = (const uint4*)embB;   // 32 uint4 per code row

    // prime the double-buffer pipeline
    stage_async_tile(embBq + (size_t)(splitBase + 0 * TILE_C) * 32 + t,
                     (unsigned)(uintptr_t)&sB[0 * 2048 + t]);
    stage_async_tile(embBq + (size_t)(splitBase + 1 * TILE_C) * 32 + t,
                     (unsigned)(uintptr_t)&sB[1 * 2048 + t]);

    for (int kt = 0; kt < TILES; ++kt) {
        const int codeBase = splitBase + kt * TILE_C;

        // prefetch e2 for this tile's 4 subtiles (covered by async wait below)
        float e2v[4];
        #pragma unroll
        for (int sub = 0; sub < 4; ++sub)
            e2v[sub] = e2[codeBase + sub * 16 + mn];

        // wait for tile kt's async copies; loads complete in order, so
        // "<= 8" is safe while tile kt+1's 8 copies are still in flight.
        if (kt + 1 < TILES) asm volatile("s_wait_asynccnt 0x8" ::: "memory");
        else                asm volatile("s_wait_asynccnt 0x0" ::: "memory");
        __syncthreads();

        const int bufBase = (kt & 1) * 2048;
        #pragma unroll
        for (int sub = 0; sub < 4; ++sub) {
            v8f acc = {0.f, 0.f, 0.f, 0.f, 0.f, 0.f, 0.f, 0.f};
            const int rb = bufBase + (sub * 16 + mn) * 32;
            #pragma unroll
            for (int f = 0; f < 8; ++f) {
                uint4 lo = sB[rb + f * 4 + half];
                uint4 hi = sB[rb + f * 4 + 2 + half];
                v8u bw = {lo.x, lo.y, lo.z, lo.w, hi.x, hi.y, hi.z, hi.w};
                v16bf b = __builtin_bit_cast(v16bf, bw);
                acc = __builtin_amdgcn_wmma_f32_16x16x32_bf16(
                    false, a[f], false, b, (short)0, acc, false, false);
            }
            const int n = codeBase + sub * 16 + mn;
            #pragma unroll
            for (int r = 0; r < 8; ++r) {
                float d = __builtin_fmaf(-2.0f, acc[r], e2v[sub]);
                if (d < minV[r]) { minV[r] = d; minIdx[r] = n; }
            }
        }
        __syncthreads();   // everyone done reading buffer (kt & 1)

        if (kt + 2 < TILES) {
            stage_async_tile(embBq + (size_t)(splitBase + (kt + 2) * TILE_C) * 32 + t,
                             (unsigned)(uintptr_t)&sB[bufBase + t]);
        }
    }

    // reduce across the 16 lanes sharing each output row (tie-break: lowest idx)
    #pragma unroll
    for (int r = 0; r < 8; ++r) {
        float v  = minV[r];
        int   id = minIdx[r];
        #pragma unroll
        for (int off = 1; off < 16; off <<= 1) {
            float ov = __shfl_xor(v, off, 32);
            int   oi = __shfl_xor(id, off, 32);
            if (ov < v || (ov == v && oi < id)) { v = ov; id = oi; }
        }
        if (mn == 0) {
            const int row = rowBase + half * 8 + r;
            pvals[(size_t)split * NPTS + row] = v;
            pidx [(size_t)split * NPTS + row] = id;
        }
    }
}

// ---------------------------------------------------------------------------
// Kernel 3: reduce K-split argmin partials, gather quantized rows, scatter
// one-hot / counts / dw, loss sum. grid = N blocks, block = 256 threads.
// ---------------------------------------------------------------------------
__global__ __launch_bounds__(256) void vq_gather_kernel(
    const float* __restrict__ x, const float* __restrict__ emb,
    const float* __restrict__ pvals, const int* __restrict__ pidx,
    float* __restrict__ quant, float* __restrict__ enc,
    float* __restrict__ newemaw, float* __restrict__ counts,
    float* __restrict__ lossacc)
{
    __shared__ float sred[8];
    __shared__ int   sid;
    const int n = blockIdx.x;
    const int t = threadIdx.x;
    if (t == 0) {
        float bv = pvals[n];
        int   bi = pidx[n];
        #pragma unroll
        for (int s = 1; s < KSPLIT; ++s) {
            float v  = pvals[(size_t)s * NPTS + n];
            int   i2 = pidx [(size_t)s * NPTS + n];
            if (v < bv || (v == bv && i2 < bi)) { bv = v; bi = i2; }
        }
        sid = bi;
    }
    __syncthreads();
    const int id = sid;

    const float xv = x[(size_t)n * DIM + t];
    const float q  = emb[(size_t)id * DIM + t];
    quant[(size_t)n * DIM + t] = q;
    atomicAdd(&newemaw[(size_t)id * DIM + t], 0.01f * xv);   // (1-DECAY)*dw
    float diff = q - xv;
    float s = diff * diff;
    #pragma unroll
    for (int off = 16; off > 0; off >>= 1) s += __shfl_xor(s, off, 32);
    if ((t & 31) == 0) sred[t >> 5] = s;
    __syncthreads();
    if (t == 0) {
        float tot = 0.f;
        #pragma unroll
        for (int w = 0; w < 8; ++w) tot += sred[w];
        atomicAdd(lossacc, tot);
        atomicAdd(&counts[id], 1.0f);
        enc[(size_t)n * NUM_E + id] = 1.0f;
    }
}

// ---------------------------------------------------------------------------
// Kernel 4: EMA cluster sizes (Laplace smoothing), perplexity, loss scalar.
// ---------------------------------------------------------------------------
__global__ __launch_bounds__(256) void vq_finalize_kernel(
    const float* __restrict__ ecsz, const float* __restrict__ counts,
    const float* __restrict__ lossacc, float* __restrict__ loss,
    float* __restrict__ perp, float* __restrict__ ecs)
{
    __shared__ float sn[8], sp[8], bc[2];
    const int t = threadIdx.x;
    float ns = 0.f, pl = 0.f;
    for (int k = t; k < NUM_E; k += 256) {
        float e = ecsz[k] * 0.99f + 0.01f * counts[k];
        ecs[k] = e;
        ns += e;
        float p = counts[k] * (1.0f / NPTS);
        pl += p * __logf(p + 1e-10f);
    }
    #pragma unroll
    for (int off = 16; off > 0; off >>= 1) {
        ns += __shfl_xor(ns, off, 32);
        pl += __shfl_xor(pl, off, 32);
    }
    if ((t & 31) == 0) { sn[t >> 5] = ns; sp[t >> 5] = pl; }
    __syncthreads();
    if (t == 0) {
        float aa = 0.f, bb = 0.f;
        #pragma unroll
        for (int w = 0; w < 8; ++w) { aa += sn[w]; bb += sp[w]; }
        bc[0] = aa; bc[1] = bb;
    }
    __syncthreads();
    const float nsum  = bc[0];
    const float scale = nsum / (nsum + (float)NUM_E * 1e-5f);
    for (int k = t; k < NUM_E; k += 256) ecs[k] = (ecs[k] + 1e-5f) * scale;
    if (t == 0) {
        loss[0] = 0.25f * lossacc[0] / ((float)NPTS * (float)DIM);
        perp[0] = __expf(-bc[1]);
    }
}

// ---------------------------------------------------------------------------
// Kernel 5: new_embedding = new_ema_w / ecs[:,None]
// ---------------------------------------------------------------------------
__global__ __launch_bounds__(256) void vq_newemb_kernel(
    const float* __restrict__ newemaw, const float* __restrict__ ecs,
    float* __restrict__ newemb)
{
    const int k = blockIdx.x;
    const int t = threadIdx.x;
    newemb[(size_t)k * DIM + t] = newemaw[(size_t)k * DIM + t] / ecs[k];
}

// ---------------------------------------------------------------------------
extern "C" void kernel_launch(void* const* d_in, const int* in_sizes, int n_in,
                              void* d_out, int out_size, void* d_ws, size_t ws_size,
                              hipStream_t stream) {
    (void)in_sizes; (void)n_in; (void)out_size; (void)ws_size;
    const float* x    = (const float*)d_in[0];   // inputs            [N,D]
    const float* emb  = (const float*)d_in[1];   // embedding         [K,D]
    const float* ecsz = (const float*)d_in[2];   // ema_cluster_size  [K]
    const float* emaw = (const float*)d_in[3];   // ema_w             [K,D]
    float* out = (float*)d_out;

    const long long ND = (long long)NPTS * DIM;
    const long long NK = (long long)NPTS * NUM_E;
    const long long KD = (long long)NUM_E * DIM;

    // output layout: loss | quantized | perplexity | encodings |
    //                new_embedding | ecs | new_ema_w
    float* o_loss    = out;
    float* o_quant   = out + 1;
    float* o_perp    = o_quant + ND;
    float* o_enc     = o_perp + 1;
    float* o_newemb  = o_enc + NK;
    float* o_ecs     = o_newemb + KD;
    float* o_newemaw = o_ecs + NUM_E;

    // workspace layout (~4.8 MB)
    char*   ws      = (char*)d_ws;
    __bf16* embB    = (__bf16*)ws;                                   // K*D*2
    float*  e2      = (float*)(ws + KD * 2);                         // K*4
    float*  pvals   = (float*)(ws + KD * 2 + NUM_E * 4);             // KSPLIT*N*4
    int*    pidx    = (int*)  (ws + KD * 2 + NUM_E * 4
                                  + (long long)KSPLIT * NPTS * 4);   // KSPLIT*N*4
    float*  counts  = (float*)(ws + KD * 2 + NUM_E * 4
                                  + 2LL * KSPLIT * NPTS * 4);        // K*4
    float*  lossacc = counts + NUM_E;                                // 4

    hipMemsetAsync(o_enc, 0, (size_t)NK * sizeof(float), stream);
    hipMemsetAsync(counts, 0, (size_t)(NUM_E + 1) * sizeof(float), stream);

    vq_prep_kernel   <<<NUM_E, 256, 0, stream>>>(emb, emaw, embB, e2, o_newemaw);
    vq_argmin_kernel <<<dim3(NPTS / 128, KSPLIT), 256, 0, stream>>>(
                                                x, embB, e2, pvals, pidx);
    vq_gather_kernel <<<NPTS,  256, 0, stream>>>(x, emb, pvals, pidx, o_quant,
                                                 o_enc, o_newemaw, counts, lossacc);
    vq_finalize_kernel<<<1,    256, 0, stream>>>(ecsz, counts, lossacc,
                                                 o_loss, o_perp, o_ecs);
    vq_newemb_kernel <<<NUM_E, 256, 0, stream>>>(o_newemaw, o_ecs, o_newemb);
}
